// AttentionTopK_29557964931072
// MI455X (gfx1250) — matrix operand order repaired and checked by reading, not compiled, not run on hardware
//
#include <hip/hip_runtime.h>
#include <hip/hip_bf16.h>
#include <math.h>

typedef __attribute__((ext_vector_type(16))) _Float16 v16h;
typedef __attribute__((ext_vector_type(8)))  float    v8f;
typedef unsigned int __attribute__((ext_vector_type(4))) v4u;
typedef int  __attribute__((ext_vector_type(8))) v8i;
typedef int  __attribute__((ext_vector_type(4))) v4i;

#define L_DIM   512
#define D_DIM   128
#define N_DIM   16384
#define B_DIM   16
#define HID_DIM 128
#define NCLS    3
#define EPSV    1e-8f

// workspace byte offsets
#define WS_WVP   0u         // 65536 f16 (packed Wv fragments)
#define WS_WUP   131072u    // 65536 f16 (packed Wu fragments)
#define WS_ARAW  262144u    // B*N f32 raw logits
#define WS_ACOPY 1310720u   // B*N f32 scratch for top-k
#define WS_POOL  2359296u   // 16*512 f32 pooled features

// ---------------------------------------------------------------------------
// Kernel 1: repack Wv/Wu (f32 -> f16) into WMMA B-fragment lane order.
// Fragment (kb,nb) = 32x16 tile of W; lane l<16 holds K=kb*32+0..15 (pairs),
// lane l>=16 holds K=kb*32+16..31, column N = nb*16 + (l%16).
// Destination: frag*512 + lane*16 + j  (each lane's 16 halfs contiguous).
// ---------------------------------------------------------------------------
__global__ __launch_bounds__(256) void pack_weights(
    const float* __restrict__ Wv, const float* __restrict__ Wu,
    _Float16* __restrict__ wvP, _Float16* __restrict__ wuP)
{
    int tid  = blockIdx.x * blockDim.x + threadIdx.x;   // 0..131071
    int sel  = tid >> 16;                               // 0: Wv, 1: Wu
    int e    = tid & 65535;
    int frag = e >> 9;          // /512
    int r    = e & 511;
    int lane = r >> 4;
    int j    = r & 15;
    int kb = frag >> 3, nb = frag & 7;
    int n  = lane & 15;
    int kbase = (lane < 16) ? 0 : 16;
    int k = kb * 32 + kbase + j;
    const float* W = sel ? Wu : Wv;
    _Float16*    P = sel ? wuP : wvP;
    P[e] = (_Float16)W[k * D_DIM + nb * 16 + n];
}

// ---------------------------------------------------------------------------
// TDM helper: one tensor_load_to_lds moving `bytes` (multiple of 8, single
// contiguous row of 8B elements) from global to LDS byte offset ldsByteOff.
// D# packing per CDNA5 ISA (2D, groups 2/3 = 0).
// Uses the 6-arg clang-23 builtin form.
// ---------------------------------------------------------------------------
__device__ __forceinline__ void tdm_load_row(const void* gptr,
                                             unsigned ldsByteOff,
                                             unsigned bytes)
{
    const unsigned elems = bytes >> 3;            // 8-byte elements
    unsigned long long ga = (unsigned long long)(size_t)gptr;
    v4u g0;
    g0[0] = 1u;                                    // count=1, user mode, no gather
    g0[1] = ldsByteOff;                            // lds_addr
    g0[2] = (unsigned)(ga & 0xffffffffu);          // global_addr[31:0]
    g0[3] = (unsigned)((ga >> 32) & 0x1ffffffu)    // global_addr[56:32]
          | (2u << 30);                            // type = 2 (image)
    v8i g1;
    g1[0] = (int)(3u << 16);                       // wg_mask=0, data_size=3 (8B)
    g1[1] = (int)((elems & 0xffffu) << 16);        // tensor_dim0[15:0]
    g1[2] = (int)(((elems >> 16) & 0xffffu)        // tensor_dim0[31:16]
          | (1u << 16));                           // tensor_dim1 = 1 (lo16)
    g1[3] = (int)((elems & 0xffffu) << 16);        // tensor_dim1 hi=0 | tile_dim0
    g1[4] = 1;                                     // tile_dim1=1, tile_dim2=0
    g1[5] = (int)elems;                            // tensor_dim0_stride[31:0]
    g1[6] = 0;                                     // stride0 hi | stride1 lo
    g1[7] = 0;                                     // stride1 hi
    v4i z4 = {};
    v8i z8 = {};
    __builtin_amdgcn_tensor_load_to_lds(g0, g1, z4, z4, z8, 0);
}

// ---------------------------------------------------------------------------
// Kernel 2: fused gated-attention scores.
// A[b,n] = (tanh(x*Wv+bv) * sigmoid(x*Wu+bu)) . Wa + ba
// Persistent waves; 256KB LDS (both packed f16 weight matrices) filled by the
// Tensor Data Mover (tensor_load_to_lds + s_wait_tensorcnt); each wave
// processes 16 rows of N with 2x8 f32 WMMA accumulator tiles, K=512 in 16
// steps of v_wmma_f32_16x16x32_f16.  x is streamed exactly once.
// ---------------------------------------------------------------------------
__global__ __launch_bounds__(256) void gating_scores(
    const float* __restrict__ x,
    const _Float16* __restrict__ wvP, const _Float16* __restrict__ wuP,
    const float* __restrict__ bv, const float* __restrict__ bu,
    const float* __restrict__ Wa, const float* __restrict__ ba,
    float* __restrict__ Araw)
{
    extern __shared__ uint4 sW[];   // [0,8192): Wv frags, [8192,16384): Wu frags
                                    // dynamic LDS starts at LDS offset 0 here

    // Wave 0 issues two TDM descriptors (128KB each), waits on TENSORcnt,
    // then the workgroup barrier publishes the LDS contents to all waves.
    if (threadIdx.x < 32) {
        tdm_load_row(wvP, 0u,      131072u);
        tdm_load_row(wuP, 131072u, 131072u);
        __builtin_amdgcn_s_wait_tensorcnt(0);
    }
    __syncthreads();

    const int lane          = threadIdx.x & 31;
    const int waveId        = threadIdx.x >> 5;
    const int wavesPerBlock = blockDim.x >> 5;
    const int globalWave    = blockIdx.x * wavesPerBlock + waveId;
    const int numWaves      = gridDim.x * wavesPerBlock;

    const int m16   = lane & 15;
    const int koffA = (lane < 16) ? 0 : 16;  // A-fragment K offset (contiguous, matches packed B)

    // per-lane constants for the epilogue: hidden index d = nb*16 + m16
    float bvr[8], bur[8], war[8];
    #pragma unroll
    for (int nb = 0; nb < 8; ++nb) {
        int d = nb * 16 + m16;
        bvr[nb] = bv[d]; bur[nb] = bu[d]; war[nb] = Wa[d];
    }
    const float ba0 = ba[0];

    const int totalTiles = B_DIM * (N_DIM / 16);
    const v8f vzero = {};

    for (int t = globalWave; t < totalTiles; t += numWaves) {
        const int b  = t >> 10;               // / (N/16)
        const int n0 = (t & 1023) << 4;
        const float* xrow = x + ((size_t)b * N_DIM + n0 + m16) * L_DIM;

        // prefetch next tile's row for this lane (global_prefetch_b8)
        {
            int tn = t + numWaves;
            if (tn < totalTiles) {
                const float* p = x + ((size_t)(tn >> 10) * N_DIM
                                      + ((tn & 1023) << 4) + m16) * L_DIM;
                __builtin_prefetch(p, 0, 1);
            }
        }

        v8f cv[8], cu[8];
        #pragma unroll
        for (int i = 0; i < 8; ++i) { cv[i] = vzero; cu[i] = vzero; }

        #pragma unroll 4
        for (int kb = 0; kb < 16; ++kb) {
            // Build A fragment: lane<16 -> K {0..15}, lane>=16 -> {16..31}
            const float* xp = xrow + kb * 32 + koffA;
            float4 c0 = reinterpret_cast<const float4*>(xp)[0];
            float4 c1 = reinterpret_cast<const float4*>(xp)[1];
            float4 c2 = reinterpret_cast<const float4*>(xp + 8)[0];
            float4 c3 = reinterpret_cast<const float4*>(xp + 8)[1];
            v16h a;
            a[0]=(_Float16)c0.x;  a[1]=(_Float16)c0.y;  a[2]=(_Float16)c0.z;  a[3]=(_Float16)c0.w;
            a[4]=(_Float16)c1.x;  a[5]=(_Float16)c1.y;  a[6]=(_Float16)c1.z;  a[7]=(_Float16)c1.w;
            a[8]=(_Float16)c2.x;  a[9]=(_Float16)c2.y;  a[10]=(_Float16)c2.z; a[11]=(_Float16)c2.w;
            a[12]=(_Float16)c3.x; a[13]=(_Float16)c3.y; a[14]=(_Float16)c3.z; a[15]=(_Float16)c3.w;

            #pragma unroll
            for (int nb = 0; nb < 8; ++nb) {
                const int fo = ((kb * 8 + nb) * 32 + lane) * 2;   // uint4 units
                union { uint4 q[2]; v16h h; } bV, bU;
                bV.q[0] = sW[fo];          bV.q[1] = sW[fo + 1];
                bU.q[0] = sW[8192 + fo];   bU.q[1] = sW[8192 + fo + 1];
                cv[nb] = __builtin_amdgcn_wmma_f32_16x16x32_f16(
                    false, a, false, bV.h, (short)0, cv[nb], false, false);
                cu[nb] = __builtin_amdgcn_wmma_f32_16x16x32_f16(
                    false, a, false, bU.h, (short)0, cu[nb], false, false);
            }
        }

        // Epilogue: gated score = sum_d tanh(v)*sigmoid(u)*Wa[d]
        float part[8];
        #pragma unroll
        for (int r = 0; r < 8; ++r) part[r] = 0.f;
        #pragma unroll
        for (int nb = 0; nb < 8; ++nb) {
            #pragma unroll
            for (int r = 0; r < 8; ++r) {
                float vv = tanhf(cv[nb][r] + bvr[nb]);
                float uu = 1.f / (1.f + expf(-(cu[nb][r] + bur[nb])));
                part[r] += vv * uu * war[nb];
            }
        }
        // reduce over hidden dim: lanes 0..15 hold rows r, lanes 16..31 rows r+8
        #pragma unroll
        for (int r = 0; r < 8; ++r) {
            #pragma unroll
            for (int off = 8; off > 0; off >>= 1)
                part[r] += __shfl_xor(part[r], off, 16);
        }
        if (m16 == 0) {
            const int rowBase = n0 + ((lane < 16) ? 0 : 8);
            #pragma unroll
            for (int r = 0; r < 8; ++r)
                Araw[(size_t)b * N_DIM + rowBase + r] = part[r] + ba0;
        }
    }
}

// ---------------------------------------------------------------------------
// Kernel 3: softmax over N, mask, renormalize.  out = e_i*m_i / (T + eps*S)
// where S = sum e, T = sum e*m.  One 1024-thread block per batch.
// ---------------------------------------------------------------------------
__global__ __launch_bounds__(1024) void softmax_mask(
    const float* __restrict__ Araw, const float* __restrict__ mask,
    float* __restrict__ Aout, float* __restrict__ Acopy)
{
    __shared__ float red[1024];
    const int b   = blockIdx.x;
    const int tid = threadIdx.x;
    const float* a  = Araw + (size_t)b * N_DIM;
    const float* mk = mask + (size_t)b * N_DIM;
    float* cp  = Acopy + (size_t)b * N_DIM;
    float* out = Aout  + (size_t)b * N_DIM;

    float lm = -3.4e38f;
    for (int i = tid; i < N_DIM; i += 1024) lm = fmaxf(lm, a[i]);
    red[tid] = lm; __syncthreads();
    for (int s = 512; s > 0; s >>= 1) {
        if (tid < s) red[tid] = fmaxf(red[tid], red[tid + s]);
        __syncthreads();
    }
    const float mx = red[0]; __syncthreads();

    float ls = 0.f, lt = 0.f;
    for (int i = tid; i < N_DIM; i += 1024) {
        float e  = expf(a[i] - mx);
        float em = e * mk[i];
        ls += e; lt += em;
        cp[i] = em;
    }
    red[tid] = ls; __syncthreads();
    for (int s = 512; s > 0; s >>= 1) { if (tid < s) red[tid] += red[tid + s]; __syncthreads(); }
    const float S = red[0]; __syncthreads();
    red[tid] = lt; __syncthreads();
    for (int s = 512; s > 0; s >>= 1) { if (tid < s) red[tid] += red[tid + s]; __syncthreads(); }
    const float T = red[0]; __syncthreads();

    const float inv = 1.f / (T + EPSV * S);
    for (int i = tid; i < N_DIM; i += 1024) {
        float v = cp[i] * inv;
        out[i] = v;
        cp[i]  = v;
    }
}

// ---------------------------------------------------------------------------
// Kernel 4: top-k selection (iterative block argmax, index tie-break to the
// smaller index) + mean pooling of selected feature rows.
// ---------------------------------------------------------------------------
__global__ __launch_bounds__(1024) void topk_pool(
    float* __restrict__ Acopy, const float* __restrict__ x,
    const int* __restrict__ topk_ptr, float* __restrict__ pooled)
{
    __shared__ unsigned long long red[1024];
    __shared__ int selIdx[128];
    const int b   = blockIdx.x;
    const int tid = threadIdx.x;
    float* av = Acopy + (size_t)b * N_DIM;

    int k = topk_ptr[0];
    if (k > 128)  k = 128;
    if (k > N_DIM) k = N_DIM;

    for (int it = 0; it < k; ++it) {
        unsigned long long best = 0ull;
        for (int i = tid; i < N_DIM; i += 1024) {
            float v = av[i];
            if (v >= 0.f) {   // softmax outputs are non-negative; -1 marks taken
                unsigned long long key =
                    ((unsigned long long)__float_as_uint(v) << 32) |
                    (unsigned)(N_DIM - i);
                if (key > best) best = key;
            }
        }
        red[tid] = best; __syncthreads();
        for (int s = 512; s > 0; s >>= 1) {
            if (tid < s) { if (red[tid + s] > red[tid]) red[tid] = red[tid + s]; }
            __syncthreads();
        }
        if (tid == 0) {
            int idx = N_DIM - (int)(red[0] & 0xFFFFFFFFull);
            selIdx[it] = idx;
            av[idx] = -1.f;
        }
        __syncthreads();
    }

    if (tid < L_DIM) {
        float s = 0.f;
        for (int j = 0; j < k; ++j)
            s += x[((size_t)b * N_DIM + selIdx[j]) * L_DIM + tid];
        pooled[b * L_DIM + tid] = s / (float)k;
    }
}

// ---------------------------------------------------------------------------
// Kernel 5: classifier head.  One wave: pooled[16,512] @ W1[512,128] via 128
// WMMAs (M=16 fits one tile), ReLU, then the 3-wide layer + argmax in VALU.
// ---------------------------------------------------------------------------
__global__ __launch_bounds__(32) void classifier(
    const float* __restrict__ pooled,
    const float* __restrict__ W1, const float* __restrict__ b1,
    const float* __restrict__ W2, const float* __restrict__ b2,
    float* __restrict__ out)
{
    __shared__ float h[16 * HID_DIM];
    const int lane   = threadIdx.x;
    const int m16    = lane & 15;
    const int koffA  = (lane < 16) ? 0 : 16;
    const int kbaseB = (lane < 16) ? 0 : 16;
    const v8f vzero = {};
    v8f c[8];
    #pragma unroll
    for (int i = 0; i < 8; ++i) c[i] = vzero;

    for (int kb = 0; kb < 16; ++kb) {
        const float* xp = pooled + m16 * L_DIM + kb * 32 + koffA;
        v16h a;
        #pragma unroll
        for (int j = 0; j < 8; ++j) a[j]     = (_Float16)xp[j];
        #pragma unroll
        for (int j = 0; j < 8; ++j) a[8 + j] = (_Float16)xp[8 + j];
        #pragma unroll
        for (int nb = 0; nb < 8; ++nb) {
            v16h bb;
            #pragma unroll
            for (int j = 0; j < 16; ++j) {
                int kk = kb * 32 + kbaseB + j;
                bb[j] = (_Float16)W1[kk * HID_DIM + nb * 16 + m16];
            }
            c[nb] = __builtin_amdgcn_wmma_f32_16x16x32_f16(
                false, a, false, bb, (short)0, c[nb], false, false);
        }
    }
    #pragma unroll
    for (int nb = 0; nb < 8; ++nb) {
        int d = nb * 16 + m16;
        float bd = b1[d];
        #pragma unroll
        for (int r = 0; r < 8; ++r) {
            int m = r + ((lane < 16) ? 0 : 8);
            float v = c[nb][r] + bd;
            h[m * HID_DIM + d] = v > 0.f ? v : 0.f;
        }
    }
    __syncthreads();
    if (lane < 16) {
        int m = lane;
        float acc[NCLS];
        #pragma unroll
        for (int cc = 0; cc < NCLS; ++cc) acc[cc] = b2[cc];
        for (int d = 0; d < HID_DIM; ++d) {
            float hv = h[m * HID_DIM + d];
            #pragma unroll
            for (int cc = 0; cc < NCLS; ++cc) acc[cc] += hv * W2[d * NCLS + cc];
        }
        int best = 0; float bm = acc[0];
        if (acc[1] > bm) { bm = acc[1]; best = 1; }
        if (acc[2] > bm) { bm = acc[2]; best = 2; }
        out[m * NCLS + 0] = acc[0];
        out[m * NCLS + 1] = acc[1];
        out[m * NCLS + 2] = acc[2];
        out[48 + m] = (float)best;   // Y_hat
    }
}

// ---------------------------------------------------------------------------
extern "C" void kernel_launch(void* const* d_in, const int* in_sizes, int n_in,
                              void* d_out, int out_size, void* d_ws, size_t ws_size,
                              hipStream_t stream) {
    (void)in_sizes; (void)n_in; (void)out_size; (void)ws_size;
    const float* x    = (const float*)d_in[0];
    const float* mask = (const float*)d_in[1];
    const float* Wv   = (const float*)d_in[2];
    const float* bv   = (const float*)d_in[3];
    const float* Wu   = (const float*)d_in[4];
    const float* bu   = (const float*)d_in[5];
    const float* Wa   = (const float*)d_in[6];
    const float* ba   = (const float*)d_in[7];
    const float* W1   = (const float*)d_in[8];
    const float* b1   = (const float*)d_in[9];
    const float* W2   = (const float*)d_in[10];
    const float* b2   = (const float*)d_in[11];
    const int*   tk   = (const int*)d_in[12];

    char* ws = (char*)d_ws;
    _Float16* wvP   = (_Float16*)(ws + WS_WVP);
    _Float16* wuP   = (_Float16*)(ws + WS_WUP);
    float*    Araw  = (float*)(ws + WS_ARAW);
    float*    Acopy = (float*)(ws + WS_ACOPY);
    float*    pool  = (float*)(ws + WS_POOL);
    float*    out   = (float*)d_out;

    pack_weights<<<512, 256, 0, stream>>>(Wv, Wu, wvP, wuP);

    const int smemBytes = 262144;   // 256KB: both packed weight matrices in LDS
    (void)hipFuncSetAttribute(reinterpret_cast<const void*>(gating_scores),
                              hipFuncAttributeMaxDynamicSharedMemorySize, smemBytes);
    gating_scores<<<256, 256, smemBytes, stream>>>(x, wvP, wuP, bv, bu, Wa, ba, Araw);

    softmax_mask<<<B_DIM, 1024, 0, stream>>>(Araw, mask, out + 64, Acopy);
    topk_pool<<<B_DIM, 1024, 0, stream>>>(Acopy, x, tk, pool);
    classifier<<<1, 32, 0, stream>>>(pool, W1, b1, W2, b2, out);
}
